// SelfAttentionWeightAllocation_55001351192998
// MI455X (gfx1250) — compile-verified, matrix operand order checked by reading, not run on hardware
//
#include <hip/hip_runtime.h>
#include <hip/hip_bf16.h>

// ---------------------------------------------------------------------------
// Problem: B=32, C=256, H=W=64, GROUPS=8 -> bg=256, cg=32, HW=4096
// ---------------------------------------------------------------------------
#define BG 256
#define CG 32
#define HH 64
#define WW 64
#define HWPIX 4096
#define EPSV 1e-5f

typedef __attribute__((ext_vector_type(16))) __bf16 v16bf;
typedef __attribute__((ext_vector_type(8)))  float  v8f;
typedef __attribute__((ext_vector_type(4)))  unsigned int v4u;
typedef __attribute__((ext_vector_type(8)))  int v8i;
typedef __attribute__((ext_vector_type(4)))  int v4i;

#if defined(__has_builtin)
#if __has_builtin(__builtin_amdgcn_tensor_load_to_lds) && __has_builtin(__builtin_amdgcn_s_wait_tensorcnt)
#define HAVE_TDM 1
#else
#define HAVE_TDM 0
#endif
#else
#define HAVE_TDM 0
#endif

union Frag {
    v16bf v;
    uint4 q[2];
};

__device__ __forceinline__ unsigned short f2bf(float f) {
    unsigned u = __float_as_uint(f);
    unsigned r = u + 0x7FFFu + ((u >> 16) & 1u);   // round-to-nearest-even
    return (unsigned short)(r >> 16);
}

__device__ __forceinline__ float sigmoidf(float x) {
    return 1.0f / (1.0f + __expf(-x));
}

__device__ __forceinline__ float blockReduceSum256(float v, float* red) {
    for (int o = 16; o > 0; o >>= 1) v += __shfl_xor(v, o, 32);
    int lane = threadIdx.x & 31, wid = threadIdx.x >> 5;
    if (lane == 0) red[wid] = v;
    __syncthreads();
    if (threadIdx.x < 8) {
        float r = red[threadIdx.x];
        for (int o = 4; o > 0; o >>= 1) r += __shfl_xor(r, o, 8);
        if (threadIdx.x == 0) red[0] = r;
    }
    __syncthreads();
    float res = red[0];
    __syncthreads();
    return res;
}

// ---------------------------------------------------------------------------
// K0: convert x (f32, [bg][c][h][w]) -> xb (bf16, [bg][h][w][c]) once.
// ---------------------------------------------------------------------------
__global__ __launch_bounds__(256) void k0_tobf16(const float* __restrict__ x,
                                                 unsigned short* __restrict__ xb) {
    int bg = blockIdx.x, h = blockIdx.y;
    __shared__ __align__(16) unsigned short sT[64 * 32];  // [w][c] = 4096 B
    int t = threadIdx.x;
    int c = t >> 3, seg = t & 7;
    const float4* p4 = (const float4*)(x + ((size_t)(bg * CG + c)) * HWPIX + h * WW + seg * 8);
    float4 v0 = p4[0], v1 = p4[1];
    unsigned short* d = &sT[(seg * 8) * 32 + c];
    d[0 * 32] = f2bf(v0.x); d[1 * 32] = f2bf(v0.y); d[2 * 32] = f2bf(v0.z); d[3 * 32] = f2bf(v0.w);
    d[4 * 32] = f2bf(v1.x); d[5 * 32] = f2bf(v1.y); d[6 * 32] = f2bf(v1.z); d[7 * 32] = f2bf(v1.w);
    __syncthreads();
    const uint4* s4 = (const uint4*)sT;
    uint4* o4 = (uint4*)(xb + ((size_t)(bg * HH + h)) * 2048);
    o4[t] = s4[t];
}

// ---------------------------------------------------------------------------
// K1: per (bg,c) row: x_h[h] = mean_w, x_w[w] = mean_h.  1 block per row.
// ---------------------------------------------------------------------------
__global__ __launch_bounds__(256) void k1_means(const float* __restrict__ x,
                                                float* __restrict__ mH,
                                                float* __restrict__ mW) {
    int row = blockIdx.x;
    const float* src = x + (size_t)row * HWPIX;
    __shared__ float cpart[256 * 16];
    int t = threadIdx.x;
    int h = t >> 2, seg = t & 3;
    const float4* p4 = (const float4*)(src + h * WW + seg * 16);
    float rp = 0.f;
    for (int j = 0; j < 4; j++) {
        float4 v = p4[j];
        cpart[t * 16 + j * 4 + 0] = v.x;
        cpart[t * 16 + j * 4 + 1] = v.y;
        cpart[t * 16 + j * 4 + 2] = v.z;
        cpart[t * 16 + j * 4 + 3] = v.w;
        rp += v.x + v.y + v.z + v.w;
    }
    rp += __shfl_down(rp, 2, 4);
    rp += __shfl_down(rp, 1, 4);
    if (seg == 0) mH[row * HH + h] = rp * (1.f / 64.f);
    __syncthreads();
    if (t < 64) {
        int sg = t >> 4, j = t & 15;
        float s = 0.f;
        for (int hh = 0; hh < 64; hh++) s += cpart[(hh * 4 + sg) * 16 + j];
        mW[row * WW + t] = s * (1.f / 64.f);
    }
}

// ---------------------------------------------------------------------------
// K2: hw = w1 @ concat(x_h, x_w) + b1 ; sigmoid -> sh, sw.  1 block per bg.
// ---------------------------------------------------------------------------
__global__ __launch_bounds__(128) void k2_gate(const float* __restrict__ w1,
                                               const float* __restrict__ b1,
                                               const float* __restrict__ mH,
                                               const float* __restrict__ mW,
                                               float* __restrict__ sh,
                                               float* __restrict__ sw) {
    int bg = blockIdx.x;
    __shared__ float V[32 * 128];
    __shared__ float Wm[32 * 32];
    __shared__ float Bv[32];
    int t = threadIdx.x;
    for (int idx = t; idx < 32 * 64; idx += 128) {
        int i = idx >> 6, k = idx & 63;
        V[i * 128 + k]      = mH[(bg * CG + i) * HH + k];
        V[i * 128 + 64 + k] = mW[(bg * CG + i) * WW + k];
    }
    for (int idx = t; idx < 1024; idx += 128) Wm[idx] = w1[idx];
    if (t < 32) Bv[t] = b1[t];
    __syncthreads();
    int k = t;
    for (int o = 0; o < 32; o++) {
        float acc = Bv[o];
        for (int i = 0; i < 32; i++) acc += Wm[o * 32 + i] * V[i * 128 + k];
        float s = sigmoidf(acc);
        if (k < 64) sh[(bg * CG + o) * HH + k] = s;
        else        sw[(bg * CG + o) * WW + (k - 64)] = s;
    }
}

// ---------------------------------------------------------------------------
// K3: gated = gx*sig_h*sig_w; per-row instance norm -> x1; row mean of x1.
// ---------------------------------------------------------------------------
__global__ __launch_bounds__(256) void k3_norm(const float* __restrict__ x,
                                               const float* __restrict__ sh,
                                               const float* __restrict__ sw,
                                               const float* __restrict__ gn_w,
                                               const float* __restrict__ gn_b,
                                               float* __restrict__ x1,
                                               float* __restrict__ meanX1) {
    int row = blockIdx.x;
    int c = row & 31;
    const float* src = x + (size_t)row * HWPIX;
    float* dst = x1 + (size_t)row * HWPIX;
    const float* shp = sh + row * HH;
    const float* swp = sw + row * WW;
    __shared__ float red[8];
    int t = threadIdx.x;
    int base = t * 16;
    int h = base >> 6, w0 = base & 63;
    float s_h = shp[h];
    float g[16];
    const float4* p4 = (const float4*)(src + base);
    float sum = 0.f, ss = 0.f;
    for (int j = 0; j < 4; j++) {
        float4 v = p4[j];
        float a0 = v.x * s_h * swp[w0 + j * 4 + 0];
        float a1 = v.y * s_h * swp[w0 + j * 4 + 1];
        float a2 = v.z * s_h * swp[w0 + j * 4 + 2];
        float a3 = v.w * s_h * swp[w0 + j * 4 + 3];
        g[j * 4 + 0] = a0; g[j * 4 + 1] = a1; g[j * 4 + 2] = a2; g[j * 4 + 3] = a3;
        sum += a0 + a1 + a2 + a3;
        ss  += a0 * a0 + a1 * a1 + a2 * a2 + a3 * a3;
    }
    float S  = blockReduceSum256(sum, red);
    float SS = blockReduceSum256(ss, red);
    float mu  = S * (1.f / 4096.f);
    float var = SS * (1.f / 4096.f) - mu * mu;
    float rs = rsqrtf(var + EPSV);
    float gw = gn_w[c], gb = gn_b[c];
    float s1 = 0.f;
    for (int j = 0; j < 16; j++) {
        float v = (g[j] - mu) * rs * gw + gb;
        s1 += v;
        dst[base + j] = v;
    }
    float S1 = blockReduceSum256(s1, red);
    if (t == 0) meanX1[row] = S1 * (1.f / 4096.f);
}

// ---------------------------------------------------------------------------
// K4: grouped 3x3 conv via WMMA bf16 (f32 acc).
//     Block = (bg, hBlock of 8 output rows). 256 threads = 8 waves.
//     LDS image: 10 rows x 66 cols x 32 ch (zero halo columns 0 and 65, row
//     stride 4224 B).  Per-row TDM tiles (strided dest can't use TDM pad,
//     interval caps at 256 dwords) -> 10 TENSOR_LOAD_TO_LDS + one
//     s_wait_tensorcnt 0.  Inner loop is branch-free: 2 ds_load_b128 (A) + 2 (B) +
//     v_wmma per tap; A fragments are re-read from sW each tap.
// ---------------------------------------------------------------------------
#define LROW 2112          // LDS row stride in ushorts (66 * 32)

__global__ __launch_bounds__(256) void k4_conv(const unsigned short* __restrict__ xb,
                                               const float* __restrict__ w3,
                                               const float* __restrict__ b3,
                                               float* __restrict__ x2) {
    __shared__ __align__(16) unsigned short sImg[10 * LROW];   // 42240 B
    __shared__ __align__(16) unsigned short sW[9 * 32 * 32];   // 18432 B
    int bg = blockIdx.x;
    int hb = blockIdx.y;
    int t = threadIdx.x;
    int wave = t >> 5, lane = t & 31;

    // weights -> LDS as bf16 (tap-major: sW[tap][o][i])
    for (int idx = t; idx < 9 * 32 * 32; idx += 256) {
        int tap = idx >> 10, rem = idx & 1023, o = rem >> 5, i = rem & 31;
        sW[idx] = f2bf(w3[(o * 32 + i) * 9 + tap]);
    }

    // zero halo columns (col 0 and col 65) of all 10 rows: 640 entries
    for (int i = t; i < 640; i += 256) {
        int r = i >> 6;                 // 64 halo entries per row
        int rem = i & 63;
        int col = (rem & 32) ? 65 : 0;
        int c = rem & 31;
        sImg[r * LROW + col * 32 + c] = 0;
    }

    // LDS row r <-> global h = hb*8 + r - 1 ; clamp at image borders
    int h0 = hb * 8 - 1;
    int r0 = 0, nrows = 10;
    if (hb == 0) {
        r0 = 1; nrows = 9; h0 = 0;
        for (int i = t; i < LROW; i += 256) sImg[i] = 0;            // row 0
    }
    if (hb == 7) {
        nrows = 9;
        for (int i = t; i < LROW; i += 256) sImg[9 * LROW + i] = 0; // row 9
    }

#if HAVE_TDM
    if (wave == 0) {
        for (int rr = 0; rr < nrows; rr++) {
            unsigned long long gaddr =
                (unsigned long long)(uintptr_t)(xb + ((size_t)(bg * HH + h0 + rr)) * 2048);
            unsigned ldsOff = (unsigned)(uintptr_t)(&sImg[((r0 + rr) * 66 + 1) * 32]);
            // D# group 0: count=1 | lds_addr | global_addr[56:0] | type=2
            v4u g0;
            g0.x = 1u;
            g0.y = ldsOff;
            g0.z = (unsigned)gaddr;
            g0.w = ((unsigned)((gaddr >> 32) & 0x01FFFFFFull)) | (2u << 30);
            // D# group 1: data_size=2B; 1-row tile of 2048 elements
            v8i g1;
            g1[0] = 0x00010000;           // [17:16] data_size = 1 (2 bytes)
            g1[1] = (int)(2048u << 16);   // tensor_dim0[15:0] = 2048
            g1[2] = (int)(1u << 16);      // tensor_dim1[15:0] = 1
            g1[3] = (int)(2048u << 16);   // tile_dim0 = 2048
            g1[4] = 1;                    // tile_dim1 = 1
            g1[5] = 2048;                 // tensor_dim0_stride low 32
            g1[6] = 0;
            g1[7] = 0;
            v4i z4 = {0, 0, 0, 0};
#if __clang_major__ >= 23
            v8i z8 = {0, 0, 0, 0, 0, 0, 0, 0};
            __builtin_amdgcn_tensor_load_to_lds(g0, g1, z4, z4, z8, 0);
#else
            __builtin_amdgcn_tensor_load_to_lds(g0, g1, z4, z4, 0);
#endif
        }
    }
    __builtin_amdgcn_s_wait_tensorcnt(0);
#else
    {
        const uint4* gsrc = (const uint4*)(xb + ((size_t)(bg * HH + h0)) * 2048);
        for (int i = t; i < nrows * 256; i += 256) {
            int rr = i >> 8, j = i & 255;
            uint4* ldst = (uint4*)&sImg[((r0 + rr) * 66 + 1) * 32];
            ldst[j] = gsrc[rr * 256 + j];
        }
    }
#endif
    __syncthreads();

    int n = lane & 15, half = lane >> 4, base0 = half * 8;

    for (int tt = 0; tt < 8; tt++) {
        int tile = wave * 8 + tt;          // 64 tiles per block
        int mo = tile >> 5;
        int rem = tile & 31;
        int hr = rem >> 2;
        int wt = rem & 3;
        v8f acc = {};
        int oA = mo * 16 + n;              // A-fragment row (output channel)
        for (int tap = 0; tap < 9; tap++) {
            int dy = tap / 3;
            int dx = tap - dy * 3;
            Frag a, b;
            const uint4* pa = (const uint4*)&sW[tap * 1024 + oA * 32 + base0];
            a.q[0] = pa[0];
            a.q[1] = pa[2];
            int lr = hr + dy;                 // LDS row
            int lc = wt * 16 + n + dx;        // LDS column 0..65 (always valid)
            const uint4* pb = (const uint4*)&sImg[lr * LROW + lc * 32 + base0];
            b.q[0] = pb[0];
            b.q[1] = pb[2];
            acc = __builtin_amdgcn_wmma_f32_16x16x32_bf16(
                false, a.v, false, b.v, (short)0, acc, false, false);
        }
        int hgl = hb * 8 + hr;
        int wcol = wt * 16 + n;
        for (int p = 0; p < 8; p++) {
            int o = mo * 16 + p + 8 * half;
            float val = acc[p] + b3[o];
            x2[((size_t)(bg * CG + o)) * HWPIX + hgl * WW + wcol] = val;
        }
    }
}

// ---------------------------------------------------------------------------
// K4b: per-bg channel means of x2 and both softmaxes.  1 block per bg.
// ---------------------------------------------------------------------------
__global__ __launch_bounds__(256) void k4b_softmax(const float* __restrict__ x2,
                                                   const float* __restrict__ meanX1,
                                                   float* __restrict__ px1,
                                                   float* __restrict__ px2) {
    int bg = blockIdx.x;
    int t = threadIdx.x;
    __shared__ float mean2[32];
    __shared__ float red[8];
    for (int c = 0; c < 32; c++) {
        const float* src = x2 + ((size_t)(bg * CG + c)) * HWPIX;
        float s = 0.f;
        for (int j = t; j < HWPIX; j += 256) s += src[j];
        s = blockReduceSum256(s, red);
        if (t == 0) mean2[c] = s * (1.f / 4096.f);
        __syncthreads();
    }
    if (t < 32) {
        float v = mean2[t];
        float mx = v;
        for (int o = 16; o > 0; o >>= 1) mx = fmaxf(mx, __shfl_xor(mx, o, 32));
        float e = __expf(v - mx);
        float se = e;
        for (int o = 16; o > 0; o >>= 1) se += __shfl_xor(se, o, 32);
        px2[bg * CG + t] = e / se;
        float v1 = meanX1[bg * CG + t];
        float mx1 = v1;
        for (int o = 16; o > 0; o >>= 1) mx1 = fmaxf(mx1, __shfl_xor(mx1, o, 32));
        float e1 = __expf(v1 - mx1);
        float se1 = e1;
        for (int o = 16; o > 0; o >>= 1) se1 += __shfl_xor(se1, o, 32);
        px1[bg * CG + t] = e1 / se1;
    }
}

// ---------------------------------------------------------------------------
// K5: weights[p] = sum_c px1[c]*x2[c,p] + px2[c]*x1[c,p];
//     out = x*(1+sigmoid(weights)).  Block = (bg, 256-pixel chunk).
// ---------------------------------------------------------------------------
__global__ __launch_bounds__(256) void k5_final(const float* __restrict__ x,
                                                const float* __restrict__ x1,
                                                const float* __restrict__ x2,
                                                const float* __restrict__ px1,
                                                const float* __restrict__ px2,
                                                float* __restrict__ out) {
    int bg = blockIdx.x;
    int chunk = blockIdx.y;
    int t = threadIdx.x;
    int p = chunk * 256 + t;
    __shared__ float a1[32], a2[32];
    if (t < 32) {
        a1[t] = px1[bg * CG + t];
        a2[t] = px2[bg * CG + t];
    }
    __syncthreads();
    size_t base = (size_t)bg * CG * HWPIX + p;
    __builtin_prefetch(&x2[base], 0, 0);
    __builtin_prefetch(&x1[base], 0, 0);
    float wsum = 0.f;
    for (int c = 0; c < 32; c++) {
        size_t e = base + (size_t)c * HWPIX;
        wsum += a1[c] * x2[e] + a2[c] * x1[e];
    }
    float s = sigmoidf(wsum);
    for (int c = 0; c < 32; c++) {
        size_t e = base + (size_t)c * HWPIX;
        out[e] = x[e] * (1.f + s);
    }
}

// ---------------------------------------------------------------------------
// launch
// ---------------------------------------------------------------------------
extern "C" void kernel_launch(void* const* d_in, const int* in_sizes, int n_in,
                              void* d_out, int out_size, void* d_ws, size_t ws_size,
                              hipStream_t stream) {
    const float* x    = (const float*)d_in[0];
    const float* w1   = (const float*)d_in[1];
    const float* b1   = (const float*)d_in[2];
    const float* w3   = (const float*)d_in[3];
    const float* b3   = (const float*)d_in[4];
    const float* gn_w = (const float*)d_in[5];
    const float* gn_b = (const float*)d_in[6];
    float* out = (float*)d_out;

    const size_t NTOT = (size_t)BG * CG * HWPIX;   // 33,554,432
    float* ws = (float*)d_ws;
    float* x1     = ws;
    float* x2     = x1 + NTOT;
    float* mH     = x2 + NTOT;
    float* mW     = mH + (size_t)BG * CG * HH;
    float* sh     = mW + (size_t)BG * CG * WW;
    float* sw     = sh + (size_t)BG * CG * HH;
    float* meanX1 = sw + (size_t)BG * CG * WW;
    float* px1    = meanX1 + (size_t)BG * CG;
    float* px2    = px1 + (size_t)BG * CG;
    unsigned short* xb = (unsigned short*)(px2 + (size_t)BG * CG);  // NTOT bf16

    k0_tobf16<<<dim3(BG, HH), 256, 0, stream>>>(x, xb);
    k1_means<<<BG * CG, 256, 0, stream>>>(x, mH, mW);
    k2_gate<<<BG, 128, 0, stream>>>(w1, b1, mH, mW, sh, sw);
    k3_norm<<<BG * CG, 256, 0, stream>>>(x, sh, sw, gn_w, gn_b, x1, meanX1);
    k4_conv<<<dim3(BG, 8), 256, 0, stream>>>(xb, w3, b3, x2);
    k4b_softmax<<<BG, 256, 0, stream>>>(x2, meanX1, px1, px2);
    k5_final<<<dim3(BG, 16), 256, 0, stream>>>(x, x1, x2, px1, px2, out);
}
